// GATNetwork_30142080483760
// MI455X (gfx1250) — matrix operand order repaired and checked by reading, not compiled, not run on hardware
//
#include <hip/hip_runtime.h>
#include <hip/hip_bf16.h>
#include <stdint.h>

typedef __attribute__((ext_vector_type(16))) __bf16 v16bf;
typedef __attribute__((ext_vector_type(8)))  float  v8f;

#define NEG_ATTN 0.2f
#define NEG_ACT  0.01f

// ---------- helpers ----------
__device__ inline __bf16 f2bf(float f) {
  unsigned u = __float_as_uint(f);
  u += 0x7FFFu + ((u >> 16) & 1u);              // round-to-nearest-even
  unsigned short s = (unsigned short)(u >> 16);
  return __builtin_bit_cast(__bf16, s);
}
// order-preserving float<->uint map (for atomicMax on floats incl. negatives)
__device__ inline unsigned ford(float f) {
  unsigned u = __float_as_uint(f);
  return (u & 0x80000000u) ? ~u : (u | 0x80000000u);
}
__device__ inline float unford(unsigned u) {
  return __uint_as_float((u & 0x80000000u) ? (u & 0x7FFFFFFFu) : ~u);
}
__device__ inline float lrelu(float v, float s) { return v > 0.f ? v : v * s; }

// ---------- init ----------
__global__ void fill_f32(float* p, float v, size_t n) {
  size_t i = (size_t)blockIdx.x * blockDim.x + threadIdx.x;
  if (i < n) p[i] = v;
}
__global__ void fill_u32(unsigned* p, unsigned v, size_t n) {
  size_t i = (size_t)blockIdx.x * blockDim.x + threadIdx.x;
  if (i < n) p[i] = v;
}

// ---------- W[K,Nc] f32 -> Wt[Nc,K] bf16 ----------
__global__ void convert_transpose_w(const float* __restrict__ W, __bf16* __restrict__ Wt,
                                    int K, int Nc) {
  int i = blockIdx.x * blockDim.x + threadIdx.x;
  if (i < K * Nc) {
    int k = i / Nc, n = i - k * Nc;
    Wt[(size_t)n * K + k] = f2bf(W[i]);
  }
}

// ---------- WMMA GEMM: C[M,Nc] = A[M,K] x W (Wt = bf16 [Nc,K]) ----------
// block: 256 thr = 8 waves; block tile 32(M) x 256(N); wave -> 16x64 strip (4 WMMAs)
// A tile staged f32 via async global->LDS (double buffered), converted to bf16 in VALU.
__global__ __launch_bounds__(256)
void gemm_bf16_wmma(const float* __restrict__ A, const __bf16* __restrict__ Wt,
                    float* __restrict__ C, int M, int K, int Nc) {
  __shared__ __align__(16) float sA[2][32 * 32];   // 2 x (32 rows x 32 K) f32 = 8KB
  const int tid  = threadIdx.x;
  const int lane = tid & 31;
  const int wave = tid >> 5;
  const int mbase = blockIdx.x * 32;
  const int msub  = wave >> 2;                     // 0/1: which 16-row half
  const int nbase = blockIdx.y * 256 + (wave & 3) * 64;
  const int mrow  = (msub << 4) + (lane & 15);     // row within 32-row tile
  const int khi   = lane >> 4;                     // half-wave selector

  // staging coords: thread -> (row, 4 floats); 256 thr x 16B = 4KB tile
  const int srow = tid >> 3;
  const int scol = (tid & 7) << 2;
  int gsrow = mbase + srow;
  if (gsrow > M - 1) gsrow = M - 1;                // clamp, never stored
  const float* gbase = A + (size_t)gsrow * K + scol;

  const int nk = K >> 5;
  v8f acc0 = {}, acc1 = {}, acc2 = {}, acc3 = {};

  auto issue_async = [&](int buf, int ki) {
    unsigned lds = (unsigned)(uintptr_t)&sA[buf][srow * 32 + scol];
    const float* g = gbase + (ki << 5);
    asm volatile("global_load_async_to_lds_b128 %0, %1, off"
                 :: "v"(lds), "v"(g) : "memory");
  };

  issue_async(0, 0);
  for (int i = 0; i < nk; ++i) {
    asm volatile("s_wait_asynccnt 0x0" ::: "memory");
    __syncthreads();
    if (i + 1 < nk) issue_async((i + 1) & 1, i + 1);

    const float* s = &sA[i & 1][0];
    // A fragment: lane(khi) holds K {khi*16..+16} contiguous (matches B)
    v16bf afrag;
    {
      const float* p0 = s + mrow * 32 + khi * 16;
#pragma unroll
      for (int e = 0; e < 8; ++e) {
        afrag[e]     = f2bf(p0[e]);
        afrag[8 + e] = f2bf(p0[8 + e]);
      }
    }
    // B fragments: columns nbase+(lane&15)+{0,16,32,48}; K = k0 + khi*16 .. +16 contiguous
    const int k0 = i << 5;
    const size_t bofs = (size_t)k0 + khi * 16;
    const int ncol = nbase + (lane & 15);
    v16bf bf0 = *(const v16bf*)&Wt[(size_t)(ncol)      * K + bofs];
    v16bf bf1 = *(const v16bf*)&Wt[(size_t)(ncol + 16) * K + bofs];
    v16bf bf2 = *(const v16bf*)&Wt[(size_t)(ncol + 32) * K + bofs];
    v16bf bf3 = *(const v16bf*)&Wt[(size_t)(ncol + 48) * K + bofs];

    acc0 = __builtin_amdgcn_wmma_f32_16x16x32_bf16(false, afrag, false, bf0,
                                                   (short)0, acc0, false, false);
    acc1 = __builtin_amdgcn_wmma_f32_16x16x32_bf16(false, afrag, false, bf1,
                                                   (short)0, acc1, false, false);
    acc2 = __builtin_amdgcn_wmma_f32_16x16x32_bf16(false, afrag, false, bf2,
                                                   (short)0, acc2, false, false);
    acc3 = __builtin_amdgcn_wmma_f32_16x16x32_bf16(false, afrag, false, bf3,
                                                   (short)0, acc3, false, false);
  }

  // C/D layout: VGPR r -> row M = r + 8*khi (within 16-row tile), col N = lane&15
  const int n = nbase + (lane & 15);
  const int gm0 = mbase + (msub << 4) + (khi << 3);
  if (mbase + 32 <= M) {                           // fast path: no bounds checks
#pragma unroll
    for (int r = 0; r < 8; ++r) {
      float* cp = C + (size_t)(gm0 + r) * Nc + n;
      cp[0] = acc0[r]; cp[16] = acc1[r]; cp[32] = acc2[r]; cp[48] = acc3[r];
    }
  } else {
#pragma unroll
    for (int r = 0; r < 8; ++r) {
      if (gm0 + r < M) {
        float* cp = C + (size_t)(gm0 + r) * Nc + n;
        cp[0] = acc0[r]; cp[16] = acc1[r]; cp[32] = acc2[r]; cp[48] = acc3[r];
      }
    }
  }
}

// ---------- el/er: wave per (n,h); el[n,h] = sum_d feat[n,h,d]*al[h,d] ----------
__global__ __launch_bounds__(256)
void compute_elr(const float* __restrict__ feat, const float* __restrict__ al,
                 const float* __restrict__ ar, float* __restrict__ el,
                 float* __restrict__ er, int N, int H, int D) {
  int gw = (int)(((size_t)blockIdx.x * blockDim.x + threadIdx.x) >> 5);
  int lane = threadIdx.x & 31;
  if (gw >= N * H) return;
  int h = gw % H;
  const float* f = feat + (size_t)gw * D;
  const float* a = al + h * D;
  const float* b = ar + h * D;
  float sl = 0.f, sr = 0.f;
  for (int d = lane; d < D; d += 32) { float v = f[d]; sl += v * a[d]; sr += v * b[d]; }
#pragma unroll
  for (int off = 16; off > 0; off >>= 1) {
    sl += __shfl_down(sl, off, 32);
    sr += __shfl_down(sr, off, 32);
  }
  if (lane == 0) { el[gw] = sl; er[gw] = sr; }
}

// ---------- edge pass 1: segment max ----------
__global__ void edge_max(const int* __restrict__ src, const int* __restrict__ dst,
                         const float* __restrict__ el, const float* __restrict__ er,
                         unsigned* __restrict__ mOrd, int E, int H) {
  int i = blockIdx.x * blockDim.x + threadIdx.x;
  if (i >= E * H) return;
  int e = i / H, h = i - e * H;
  int s = src[e], d = dst[e];
  float v = lrelu(el[s * H + h] + er[d * H + h], NEG_ATTN);
  atomicMax(&mOrd[d * H + h], ford(v));
}

// ---------- edge pass 2: segment sum of exp ----------
__global__ void edge_sum(const int* __restrict__ src, const int* __restrict__ dst,
                         const float* __restrict__ el, const float* __restrict__ er,
                         const unsigned* __restrict__ mOrd, float* __restrict__ ssum,
                         int E, int H) {
  int i = blockIdx.x * blockDim.x + threadIdx.x;
  if (i >= E * H) return;
  int e = i / H, h = i - e * H;
  int s = src[e], d = dst[e];
  float v = lrelu(el[s * H + h] + er[d * H + h], NEG_ATTN);
  float m = unford(mOrd[d * H + h]);
  atomicAdd(&ssum[d * H + h], __expf(v - m));
}

// ---------- edge pass 3: aggregate (wave per (edge, head), float4 gathers) ----------
__global__ __launch_bounds__(256)
void edge_agg(const int* __restrict__ src, const int* __restrict__ dst,
              const float* __restrict__ el, const float* __restrict__ er,
              const unsigned* __restrict__ mOrd, const float* __restrict__ ssum,
              const float* __restrict__ feat, float* __restrict__ agg,
              int E, int H, int D) {
  int gw = (int)(((size_t)blockIdx.x * blockDim.x + threadIdx.x) >> 5);
  int lane = threadIdx.x & 31;
  if (gw >= E * H) return;
  int e = gw / H, h = gw - e * H;
  int s = src[e], d = dst[e];
  float v = lrelu(el[s * H + h] + er[d * H + h], NEG_ATTN);
  float m = unford(mOrd[d * H + h]);
  float alpha = __expf(v - m) / ssum[d * H + h];
  const float* f = feat + ((size_t)s * H + h) * D;
  float* o = agg + ((size_t)d * H + h) * D;
  for (int dd = lane * 4; dd < D; dd += 128) {
    float4 v4 = *(const float4*)(f + dd);
    atomicAdd(o + dd + 0, v4.x * alpha);
    atomicAdd(o + dd + 1, v4.y * alpha);
    atomicAdd(o + dd + 2, v4.z * alpha);
    atomicAdd(o + dd + 3, v4.w * alpha);
  }
}

// ---------- bias + optional leaky_relu ----------
__global__ void bias_act(const float* __restrict__ agg, const float* __restrict__ b,
                         float* __restrict__ out, size_t n, int HD, int activate) {
  size_t i = (size_t)blockIdx.x * blockDim.x + threadIdx.x;
  if (i >= n) return;
  float v = agg[i] + b[i % HD];
  out[i] = activate ? lrelu(v, NEG_ACT) : v;
}

// ---------- layer 2: matvec (h[N,256] x w[256]) via one wave per node ----------
__global__ __launch_bounds__(256)
void matvec256(const float* __restrict__ h, const float* __restrict__ w,
               float* __restrict__ out, int N) {
  int gw = (int)(((size_t)blockIdx.x * blockDim.x + threadIdx.x) >> 5);
  int lane = threadIdx.x & 31;
  if (gw >= N) return;
  const float* row = h + (size_t)gw * 256;
  float s = 0.f;
  for (int k = lane; k < 256; k += 32) s += row[k] * w[k];
#pragma unroll
  for (int off = 16; off > 0; off >>= 1) s += __shfl_down(s, off, 32);
  if (lane == 0) out[gw] = s;
}

// ---------- scalar-head (H=1, D=1) edge passes for layer 2 ----------
__global__ void edge_max1(const int* __restrict__ src, const int* __restrict__ dst,
                          const float* __restrict__ f2, const float* __restrict__ al,
                          const float* __restrict__ ar, unsigned* __restrict__ mOrd, int E) {
  int e = blockIdx.x * blockDim.x + threadIdx.x;
  if (e >= E) return;
  int s = src[e], d = dst[e];
  float v = lrelu(f2[s] * al[0] + f2[d] * ar[0], NEG_ATTN);
  atomicMax(&mOrd[d], ford(v));
}
__global__ void edge_sum1(const int* __restrict__ src, const int* __restrict__ dst,
                          const float* __restrict__ f2, const float* __restrict__ al,
                          const float* __restrict__ ar, const unsigned* __restrict__ mOrd,
                          float* __restrict__ ssum, int E) {
  int e = blockIdx.x * blockDim.x + threadIdx.x;
  if (e >= E) return;
  int s = src[e], d = dst[e];
  float v = lrelu(f2[s] * al[0] + f2[d] * ar[0], NEG_ATTN);
  atomicAdd(&ssum[d], __expf(v - unford(mOrd[d])));
}
__global__ void edge_agg1(const int* __restrict__ src, const int* __restrict__ dst,
                          const float* __restrict__ f2, const float* __restrict__ al,
                          const float* __restrict__ ar, const unsigned* __restrict__ mOrd,
                          const float* __restrict__ ssum, float* __restrict__ agg, int E) {
  int e = blockIdx.x * blockDim.x + threadIdx.x;
  if (e >= E) return;
  int s = src[e], d = dst[e];
  float v = lrelu(f2[s] * al[0] + f2[d] * ar[0], NEG_ATTN);
  float alpha = __expf(v - unford(mOrd[d])) / ssum[d];
  atomicAdd(&agg[d], f2[s] * alpha);
}
__global__ void final_add(const float* __restrict__ agg, const float* __restrict__ b,
                          float* __restrict__ out, int N) {
  int n = blockIdx.x * blockDim.x + threadIdx.x;
  if (n < N) out[n] = agg[n] + b[0];
}

// ============================ launcher ============================
static inline size_t alignup(size_t x) { return (x + 255) & ~(size_t)255; }

extern "C" void kernel_launch(void* const* d_in, const int* in_sizes, int n_in,
                              void* d_out, int out_size, void* d_ws, size_t ws_size,
                              hipStream_t stream) {
  const float* features = (const float*)d_in[0];
  const int*   src      = (const int*)d_in[1];
  const int*   dst      = (const int*)d_in[2];
  const float* W0  = (const float*)d_in[3];
  const float* al0 = (const float*)d_in[4];
  const float* ar0 = (const float*)d_in[5];
  const float* b0  = (const float*)d_in[6];
  const float* W1  = (const float*)d_in[7];
  const float* al1 = (const float*)d_in[8];
  const float* ar1 = (const float*)d_in[9];
  const float* b1  = (const float*)d_in[10];
  const float* W2  = (const float*)d_in[11];
  const float* al2 = (const float*)d_in[12];
  const float* ar2 = (const float*)d_in[13];
  const float* b2  = (const float*)d_in[14];

  const int IN = 128, H = 2, D = 128, HD = 256;
  const int N = in_sizes[0] / IN;
  const int E = in_sizes[1];

  // workspace carve
  char* w = (char*)d_ws;
  __bf16* Wt0 = (__bf16*)w;  w += alignup((size_t)IN * HD * sizeof(__bf16));
  __bf16* Wt1 = (__bf16*)w;  w += alignup((size_t)HD * HD * sizeof(__bf16));
  float* feat = (float*)w;   w += alignup((size_t)N * HD * sizeof(float));
  float* agg  = (float*)w;   w += alignup((size_t)N * HD * sizeof(float));
  float* h0   = (float*)w;   w += alignup((size_t)N * HD * sizeof(float));
  float* el   = (float*)w;   w += alignup((size_t)N * H * sizeof(float));
  float* er   = (float*)w;   w += alignup((size_t)N * H * sizeof(float));
  unsigned* mOrd = (unsigned*)w; w += alignup((size_t)N * H * sizeof(unsigned));
  float* ssum = (float*)w;   w += alignup((size_t)N * H * sizeof(float));
  float* f2   = (float*)w;   w += alignup((size_t)N * sizeof(float));

  float* h1 = (float*)d_out + N;          // second tuple output [N,256]
  float* out0 = (float*)d_out;            // first tuple output [N]

  const unsigned ORD_NEG_INF = 0x007FFFFFu;  // ford(-inf)
  const int T = 256;
  auto g1 = [&](size_t n) { return (unsigned)((n + T - 1) / T); };

  convert_transpose_w<<<g1((size_t)IN * HD), T, 0, stream>>>(W0, Wt0, IN, HD);
  convert_transpose_w<<<g1((size_t)HD * HD), T, 0, stream>>>(W1, Wt1, HD, HD);

  dim3 gemmGrid((N + 31) / 32, HD / 256);
  unsigned edgeWaveBlocks = g1((size_t)E * H * 32);

  // ================= layer 0 =================
  gemm_bf16_wmma<<<gemmGrid, T, 0, stream>>>(features, Wt0, feat, N, IN, HD);
  compute_elr<<<g1((size_t)N * H * 32), T, 0, stream>>>(feat, al0, ar0, el, er, N, H, D);
  fill_u32<<<g1((size_t)N * H), T, 0, stream>>>(mOrd, ORD_NEG_INF, (size_t)N * H);
  fill_f32<<<g1((size_t)N * H), T, 0, stream>>>(ssum, 0.f, (size_t)N * H);
  fill_f32<<<g1((size_t)N * HD), T, 0, stream>>>(agg, 0.f, (size_t)N * HD);
  edge_max<<<g1((size_t)E * H), T, 0, stream>>>(src, dst, el, er, mOrd, E, H);
  edge_sum<<<g1((size_t)E * H), T, 0, stream>>>(src, dst, el, er, mOrd, ssum, E, H);
  edge_agg<<<edgeWaveBlocks, T, 0, stream>>>(src, dst, el, er, mOrd, ssum, feat, agg, E, H, D);
  bias_act<<<g1((size_t)N * HD), T, 0, stream>>>(agg, b0, h0, (size_t)N * HD, HD, 1);

  // ================= layer 1 =================
  gemm_bf16_wmma<<<gemmGrid, T, 0, stream>>>(h0, Wt1, feat, N, HD, HD);
  compute_elr<<<g1((size_t)N * H * 32), T, 0, stream>>>(feat, al1, ar1, el, er, N, H, D);
  fill_u32<<<g1((size_t)N * H), T, 0, stream>>>(mOrd, ORD_NEG_INF, (size_t)N * H);
  fill_f32<<<g1((size_t)N * H), T, 0, stream>>>(ssum, 0.f, (size_t)N * H);
  fill_f32<<<g1((size_t)N * HD), T, 0, stream>>>(agg, 0.f, (size_t)N * HD);
  edge_max<<<g1((size_t)E * H), T, 0, stream>>>(src, dst, el, er, mOrd, E, H);
  edge_sum<<<g1((size_t)E * H), T, 0, stream>>>(src, dst, el, er, mOrd, ssum, E, H);
  edge_agg<<<edgeWaveBlocks, T, 0, stream>>>(src, dst, el, er, mOrd, ssum, feat, agg, E, H, D);
  bias_act<<<g1((size_t)N * HD), T, 0, stream>>>(agg, b1, h1, (size_t)N * HD, HD, 1);

  // ================= layer 2 (single head, D=1) =================
  matvec256<<<g1((size_t)N * 32), T, 0, stream>>>(h1, W2, f2, N);
  fill_u32<<<g1((size_t)N), T, 0, stream>>>(mOrd, ORD_NEG_INF, (size_t)N);
  fill_f32<<<g1((size_t)N), T, 0, stream>>>(ssum, 0.f, (size_t)N);
  fill_f32<<<g1((size_t)N), T, 0, stream>>>(agg, 0.f, (size_t)N);
  edge_max1<<<g1((size_t)E), T, 0, stream>>>(src, dst, f2, al2, ar2, mOrd, E);
  edge_sum1<<<g1((size_t)E), T, 0, stream>>>(src, dst, f2, al2, ar2, mOrd, ssum, E);
  edge_agg1<<<g1((size_t)E), T, 0, stream>>>(src, dst, f2, al2, ar2, mOrd, ssum, agg, E);
  final_add<<<g1((size_t)N), T, 0, stream>>>(agg, b2, out0, N);

  (void)n_in; (void)out_size; (void)ws_size;
}
